// BatteriesMACE_33509334843734
// MI455X (gfx1250) — matrix operand order repaired and checked by reading, not compiled, not run on hardware
//
#include <hip/hip_runtime.h>
#include <stdint.h>

// ---------------------------------------------------------------------------
// MACE-like GNN for MI455X (gfx1250, wave32, WMMA + TDM).
//  - msg:  TDM gather-mode load of 16 src-node rows -> LDS (padded stride 514),
//          per-m 16x16x32 bf16 WMMA tensor product, straight-line f32 atomic
//          scatter into agg via 32-bit offsets from a uniform base (saddr form).
//  - lin:  dense 512x512 bf16-WMMA GEMM, fused bias + residual.
// ---------------------------------------------------------------------------

typedef __attribute__((ext_vector_type(16))) __bf16        v16bf;
typedef __attribute__((ext_vector_type(8)))  float         v8f;
typedef __attribute__((ext_vector_type(8)))  unsigned int  v8u;
typedef __attribute__((ext_vector_type(4)))  unsigned int  v4u;
typedef __attribute__((ext_vector_type(8)))  int           v8i;
typedef __attribute__((ext_vector_type(4)))  int           v4i;

#define D_FEAT 512
#define H_CH   32
#define NI_SH  16
#define ROW_STRIDE 514   // 512 dwords + 2 TDM pad dwords (1 per 256) -> bank spread

// float -> bf16 (RNE), two packed into one dword
static __device__ __forceinline__ unsigned int pack_bf2(float lo, float hi) {
    unsigned int ulo = __builtin_bit_cast(unsigned int, lo);
    unsigned int uhi = __builtin_bit_cast(unsigned int, hi);
    ulo += 0x7FFFu + ((ulo >> 16) & 1u);
    uhi += 0x7FFFu + ((uhi >> 16) & 1u);
    return (ulo >> 16) | (uhi & 0xFFFF0000u);
}

static __device__ __forceinline__ v8f wmma_bf16(v8u a, v8u b, v8f c) {
    return __builtin_amdgcn_wmma_f32_16x16x32_bf16(
        false, __builtin_bit_cast(v16bf, a),
        false, __builtin_bit_cast(v16bf, b),
        (short)0, c, false, false);
}

// A-matrix (16x32, 16-bit) K packing per lane-half (ISA 7.12.2)
static __device__ __forceinline__ int a_kbase(int i, int half) {
    return (i < 4) ? (2 * i + 8 * half) : (16 + 2 * (i - 4) + 8 * half);
}

// padded LDS index: one pad dword after every 256 dwords
static __device__ __forceinline__ int pidx(int off) { return off + (off >> 8); }

// ---------------------------------------------------------------------------
// Spherical harmonics (l<=3) per edge, with PBC minimum image.
// ---------------------------------------------------------------------------
__global__ void sh_kernel(const float* __restrict__ pos,
                          const float* __restrict__ cell,
                          const int*   __restrict__ edge,   // [2,E]
                          float*       __restrict__ Yout,   // [E,16]
                          int E) {
    int e = blockIdx.x * blockDim.x + threadIdx.x;
    if (e >= E) return;
    int s = edge[e];
    int d = edge[E + e];
    float cx = cell[0], cy = cell[4], cz = cell[8];
    float ex = pos[d * 3 + 0] - pos[s * 3 + 0];
    float ey = pos[d * 3 + 1] - pos[s * 3 + 1];
    float ez = pos[d * 3 + 2] - pos[s * 3 + 2];
    ex -= rintf(ex / cx) * cx;
    ey -= rintf(ey / cy) * cy;
    ez -= rintf(ez / cz) * cz;
    float r = sqrtf(ex * ex + ey * ey + ez * ez);
    r = fmaxf(r, 1e-8f);
    float x = ex / r, y = ey / r, z = ez / r;
    float x2 = x * x, y2 = y * y, z2 = z * z;
    float* o = Yout + (size_t)e * NI_SH;
    o[0]  = 0.28209479177387814f;
    o[1]  = 0.4886025119029199f * y;
    o[2]  = 0.4886025119029199f * z;
    o[3]  = 0.4886025119029199f * x;
    o[4]  = 1.0925484305920792f * x * y;
    o[5]  = 1.0925484305920792f * y * z;
    o[6]  = 0.31539156525252005f * (3.0f * z2 - 1.0f);
    o[7]  = 1.0925484305920792f * x * z;
    o[8]  = 0.5462742152960396f * (x2 - y2);
    o[9]  = 0.5900435899266435f * y * (3.0f * x2 - y2);
    o[10] = 2.890611442640554f * x * y * z;
    o[11] = 0.4570457994644658f * y * (5.0f * z2 - 1.0f);
    o[12] = 0.3731763325901154f * z * (5.0f * z2 - 3.0f);
    o[13] = 0.4570457994644658f * x * (5.0f * z2 - 1.0f);
    o[14] = 1.445305721320277f * z * (x2 - y2);
    o[15] = 0.5900435899266435f * x * (x2 - 3.0f * y2);
}

__global__ void embed_kernel(const float* __restrict__ emb,
                             const int*   __restrict__ an,
                             float*       __restrict__ nf,
                             int N) {
    size_t t = (size_t)blockIdx.x * blockDim.x + threadIdx.x;
    size_t total = (size_t)N * D_FEAT;
    if (t >= total) return;
    int n = (int)(t >> 9);
    int j = (int)(t & (D_FEAT - 1));
    nf[t] = emb[(size_t)an[n] * D_FEAT + j];
}

__global__ void zero_kernel(float* __restrict__ p, size_t total) {
    size_t t = (size_t)blockIdx.x * blockDim.x + threadIdx.x;
    if (t < total) p[t] = 0.0f;
}

__global__ void init_out_kernel(float* __restrict__ out,
                                const float* __restrict__ out_b, int N) {
    out[0] = (float)N * out_b[0];
}

// ---------------------------------------------------------------------------
// Message + scatter: one wave / 16 edges.
//   msg[e,d,m] = sum_c nf[src[e],c,m] * Y[e,m] * W[l(m),c,d]
// ---------------------------------------------------------------------------
__global__ __launch_bounds__(32) void msg_kernel(
    const float* __restrict__ nf,    // [N,512] (c*16+m)
    const float* __restrict__ Y,     // [E,16]
    const int*   __restrict__ edge,  // [2,E]
    const float* __restrict__ tpw,   // [4,32,32] (l,c,d)
    float*       __restrict__ agg,   // [N+1,512]  (row N = dump row)
    int E, int N) {
    __shared__ float sNF[16 * ROW_STRIDE];   // must be first LDS object
    __shared__ float sW[4 * H_CH * H_CH];
    __shared__ float sY[16 * NI_SH];

    const int lane = threadIdx.x;
    const int half = lane >> 4;
    const int l15  = lane & 15;
    const int eb   = blockIdx.x * 16;

    // Wave-uniform src indices + 32-bit dst row offsets (scalar loads).
    int srcs[16];
    unsigned rowoff[16];                       // dst_row * 512 (fits 32 bits)
#pragma unroll
    for (int e = 0; e < 16; ++e) {
        int ge = eb + e;
        srcs[e] = (ge < E) ? edge[ge] : 0;
        int dr  = (ge < E) ? edge[E + ge] : N; // dump row for tail edges
        rowoff[e] = (unsigned)dr * D_FEAT;
    }

#if __has_builtin(__builtin_amdgcn_tensor_load_to_lds)
    // ---- TDM gather-mode: 16 rows of 512 f32, LDS-padded (1 dword / 256) ----
    {
        unsigned lds_off = (unsigned)(size_t)(void*)sNF;
        unsigned long long ga = (unsigned long long)(size_t)nf;
        v4u g0 = { 0x80000001u,                 // count=1, gather_mode=1, idx16
                   lds_off,
                   (unsigned)(ga & 0xffffffffull),
                   (unsigned)((ga >> 32) & 0x01ffffffull) | (2u << 30) };
        v8i g1 = { (int)((2u << 16) | (1u << 20) | (7u << 22)), // 4B, pad 1/256
                   (int)(512u << 16),                            // tensor_dim0
                   (int)(((unsigned)N & 0xffffu) << 16),         // tensor_dim1 lo
                   (int)((((unsigned)N >> 16) & 0xffffu) | (512u << 16)), // tile_dim0
                   16,                                           // tile_dim1 = #idx
                   512,                                          // dim0 stride
                   0, 0 };
        v4i g2 = { (int)((srcs[0] & 0xffff)  | (srcs[1]  << 16)),
                   (int)((srcs[2] & 0xffff)  | (srcs[3]  << 16)),
                   (int)((srcs[4] & 0xffff)  | (srcs[5]  << 16)),
                   (int)((srcs[6] & 0xffff)  | (srcs[7]  << 16)) };
        v4i g3 = { (int)((srcs[8] & 0xffff)  | (srcs[9]  << 16)),
                   (int)((srcs[10] & 0xffff) | (srcs[11] << 16)),
                   (int)((srcs[12] & 0xffff) | (srcs[13] << 16)),
                   (int)((srcs[14] & 0xffff) | (srcs[15] << 16)) };
        v8i g4 = { 0, 0, 0, 0, 0, 0, 0, 0 };    // trailing operand (6-arg form)
        __builtin_amdgcn_tensor_load_to_lds(g0, g1, g2, g3, g4, 0);
    }
#else
    // ---- fallback: manual gather into the same padded layout ----
    for (int e = 0; e < 16; ++e) {
        const float4* p4 = (const float4*)(nf + (size_t)srcs[e] * D_FEAT);
        float* dr = &sNF[e * ROW_STRIDE];
        for (int t = lane; t < D_FEAT / 4; t += 32) {
            float4 v = p4[t];
            int pb = pidx(t * 4);
            dr[pb] = v.x; dr[pb + 1] = v.y; dr[pb + 2] = v.z; dr[pb + 3] = v.w;
        }
    }
#endif

    // Weights + Y into LDS (overlaps with the TDM transfer).
    for (int t = lane; t < 4 * H_CH * H_CH; t += 32) sW[t] = tpw[t];
    for (int t = lane; t < 16 * NI_SH; t += 32) {
        int ge = eb + (t >> 4);
        sY[t] = (ge < E) ? Y[(size_t)ge * NI_SH + (t & 15)] : 0.0f;
    }
    __syncthreads();
#if __has_builtin(__builtin_amdgcn_s_wait_tensorcnt)
    __builtin_amdgcn_s_wait_tensorcnt(0);
#endif

    const float* nfrow = &sNF[l15 * ROW_STRIDE];
    const unsigned dcol = (unsigned)(l15 * NI_SH);   // per-lane d-column offset
    const int mlo[4] = {0, 1, 4, 9};
    const int mhi[4] = {1, 4, 9, 16};

    for (int l = 0; l < 4; ++l) {
        // B fragments for this l only (16 VGPRs live, no spills).
        // B layout: lane n = lane%16, VGPR j holds K = a_kbase(j, half) (same as A).
        v8u B0, B1;
#pragma unroll
        for (int j = 0; j < 8; ++j) {
            int k0 = a_kbase(j, half);
            const float* wr = &sW[(l * H_CH + k0) * H_CH + l15];
            B0[j] = pack_bf2(wr[0], wr[H_CH]);
            B1[j] = pack_bf2(wr[16], wr[H_CH + 16]);
        }
        for (int m = mlo[l]; m < mhi[l]; ++m) {
            float ym = sY[l15 * NI_SH + m];
            v8u au;
#pragma unroll
            for (int i = 0; i < 8; ++i) {
                int c0 = a_kbase(i, half);
                au[i] = pack_bf2(nfrow[pidx(c0 * NI_SH + m)] * ym,
                                 nfrow[pidx((c0 + 1) * NI_SH + m)] * ym);
            }
            v8f z = {};
            v8f d0 = wmma_bf16(au, B0, z);   // d = 0..15
            v8f d1 = wmma_bf16(au, B1, z);   // d = 16..31
            // D layout: lane n = lane%16 (= d), VGPR i -> edge row i+8*half.
            // 32-bit element offsets from uniform base -> saddr atomic form.
            // No guards: tail edges have Y=0 -> add 0.0 into dump row N.
#pragma unroll
            for (int i = 0; i < 8; ++i) {
                unsigned off = rowoff[i + 8 * half] + dcol + (unsigned)m;
                atomicAdd(&agg[off], d0[i]);
                atomicAdd(&agg[off + 256u], d1[i]);   // (d+16)*16 = +256 elems
            }
        }
    }
}

// ---------------------------------------------------------------------------
// Dense linear with residual: nf[n,j] += sum_k agg[n,k]*W[k,j] + b[j]
// 8 waves/block; each wave owns one 16x16 tile, K=512 in 16 WMMA steps.
// ---------------------------------------------------------------------------
__global__ __launch_bounds__(256) void lin_kernel(
    const float* __restrict__ agg,   // [N(+1),512]
    const float* __restrict__ W,     // [512,512]
    const float* __restrict__ bias,  // [512]
    float*       __restrict__ nf,    // [N,512]
    int N) {
    const int lane = threadIdx.x & 31;
    const int wave = threadIdx.x >> 5;
    const int half = lane >> 4;
    const int l15  = lane & 15;
    const int n0   = blockIdx.x * 16;
    const int j0   = (blockIdx.y * 8 + wave) * 16;
    const unsigned col = (unsigned)(j0 + l15);

    int nrow = n0 + l15;
    if (nrow >= N) nrow = N - 1;
    const unsigned abase = (unsigned)nrow * D_FEAT;  // 32-bit element offsets

    v8f acc = {};
    for (int k0 = 0; k0 < D_FEAT; k0 += 32) {
        v8u au, bu;
#pragma unroll
        for (int i = 0; i < 8; ++i) {
            unsigned c0 = abase + (unsigned)(a_kbase(i, half) + k0);
            au[i] = pack_bf2(agg[c0], agg[c0 + 1]);
        }
#pragma unroll
        for (int j = 0; j < 8; ++j) {
            unsigned kk = (unsigned)(k0 + a_kbase(j, half));
            bu[j] = pack_bf2(W[kk * D_FEAT + col], W[(kk + 1) * D_FEAT + col]);
        }
        acc = wmma_bf16(au, bu, acc);
    }

    float bj = bias[col];
#pragma unroll
    for (int i = 0; i < 8; ++i) {
        int node = n0 + i + 8 * half;
        if (node < N) nf[(unsigned)node * D_FEAT + col] += acc[i] + bj;
    }
}

// ---------------------------------------------------------------------------
// energy += sum_n nf[n,:] . out_w
// ---------------------------------------------------------------------------
__global__ void energy_kernel(const float* __restrict__ nf,
                              const float* __restrict__ ow,
                              float* __restrict__ out, int N) {
    __shared__ float red[256];
    size_t total = (size_t)N * D_FEAT;
    float s = 0.0f;
    for (size_t t = (size_t)blockIdx.x * blockDim.x + threadIdx.x; t < total;
         t += (size_t)gridDim.x * blockDim.x)
        s += nf[t] * ow[t & (D_FEAT - 1)];
    red[threadIdx.x] = s;
    __syncthreads();
    for (int o = 128; o > 0; o >>= 1) {
        if (threadIdx.x < o) red[threadIdx.x] += red[threadIdx.x + o];
        __syncthreads();
    }
    if (threadIdx.x == 0) atomicAdd(out, red[0]);
}

// ---------------------------------------------------------------------------
extern "C" void kernel_launch(void* const* d_in, const int* in_sizes, int n_in,
                              void* d_out, int out_size, void* d_ws, size_t ws_size,
                              hipStream_t stream) {
    const float* pos    = (const float*)d_in[0];
    const float* cell   = (const float*)d_in[1];
    const int*   an     = (const int*)  d_in[2];
    const int*   edge   = (const int*)  d_in[3];
    const float* emb    = (const float*)d_in[4];
    const float* tp_w0  = (const float*)d_in[5];
    const float* lin_w0 = (const float*)d_in[6];
    const float* lin_b0 = (const float*)d_in[7];
    const float* tp_w1  = (const float*)d_in[8];
    const float* lin_w1 = (const float*)d_in[9];
    const float* lin_b1 = (const float*)d_in[10];
    const float* out_w  = (const float*)d_in[11];
    const float* out_b  = (const float*)d_in[12];
    float* out = (float*)d_out;

    const int N = in_sizes[0] / 3;
    const int E = in_sizes[3] / 2;

    float* nf  = (float*)d_ws;                       // [N,512]
    float* agg = nf + (size_t)N * D_FEAT;            // [N+1,512] (+dump row)
    float* Yw  = agg + (size_t)(N + 1) * D_FEAT;     // [E,16]

    const size_t nfTot  = (size_t)N * D_FEAT;
    const size_t aggTot = (size_t)(N + 1) * D_FEAT;
    const int thr = 256;

    init_out_kernel<<<1, 1, 0, stream>>>(out, out_b, N);
    sh_kernel<<<(E + thr - 1) / thr, thr, 0, stream>>>(pos, cell, edge, Yw, E);
    embed_kernel<<<(int)((nfTot + thr - 1) / thr), thr, 0, stream>>>(emb, an, nf, N);

    const float* tpw[2]  = {tp_w0, tp_w1};
    const float* linw[2] = {lin_w0, lin_w1};
    const float* linb[2] = {lin_b0, lin_b1};
    for (int layer = 0; layer < 2; ++layer) {
        zero_kernel<<<(int)((aggTot + thr - 1) / thr), thr, 0, stream>>>(agg, aggTot);
        msg_kernel<<<(E + 15) / 16, 32, 0, stream>>>(nf, Yw, edge, tpw[layer], agg, E, N);
        dim3 lg((N + 15) / 16, D_FEAT / (8 * 16));
        lin_kernel<<<lg, 256, 0, stream>>>(agg, linw[layer], linb[layer], nf, N);
    }
    energy_kernel<<<1024, 256, 0, stream>>>(nf, out_w, out, N);
}
